// SparseAxialCausalAttention_9165460209907
// MI455X (gfx1250) — matrix operand order repaired and checked by reading, not compiled, not run on hardware
//
#include <hip/hip_runtime.h>
#include <hip/hip_bf16.h>

// ---------------------------------------------------------------------------
// Sparse axial causal attention for MI455X (gfx1250), wave32 + WMMA f16.
// All matmul operands are staged so that every WMMA fragment is built from
// contiguous 128-bit loads (weights pre-transposed in the cast kernel, V in the QKV
// kernel). Exact f32 softmax via per-wave LDS scoreboards.
// ---------------------------------------------------------------------------

typedef __attribute__((ext_vector_type(16))) _Float16 v16h;
typedef __attribute__((ext_vector_type(8)))  _Float16 v8h;
typedef __attribute__((ext_vector_type(8)))  float    v8f;

#define NEGF (-3.4028234663852886e+38f)

union Frag16 { v16h v; v8h h[2]; };

// D = A(16x32) * B(32x16) + C, f32 accum
static __device__ inline v8f wmma32(v16h a, v16h b, v8f c) {
  return __builtin_amdgcn_wmma_f32_16x16x32_f16(
      /*neg_a=*/false, a, /*neg_b=*/false, b,
      /*c_mod=*/(short)0, c, /*reuse_a=*/false, /*reuse_b=*/false);
}

// A fragment, source row-major [M][K] with row pitch `pitch` (halves).
// Lane L: row = m0 + (L&15); halves 0..7 -> K = k0 + klo + i,
// halves 8..15 -> K = k0 + 16 + klo + i, klo = (L>=16)?8:0.   (ISA 7.12.2)
static __device__ inline v16h load_a_rm(const _Float16* base, int pitch,
                                        int m0, int k0, int lane) {
  int r   = lane & 15;
  int klo = (lane >> 4) << 3;
  const _Float16* p = base + (size_t)(m0 + r) * pitch + k0 + klo;
  Frag16 f;
  f.h[0] = *(const v8h*)(p);
  f.h[1] = *(const v8h*)(p + 16);
  return f.v;
}

// B fragment from the TRANSPOSE of the GEMM's B: source row-major [N][K].
// Lane L: source row = n0 + (L&15); halves 0..7 -> K = k0 + koff + i,
// halves 8..15 -> K = k0 + 16 + koff + i, koff = (L>=16)?8:0.  Two 16B loads.
static __device__ inline v16h load_bt(const _Float16* base, int pitch,
                                      int n0, int k0, int lane) {
  int nrow = n0 + (lane & 15);
  int koff = (lane >> 4) << 3;
  const _Float16* p = base + (size_t)nrow * pitch + k0 + koff;
  Frag16 f;
  f.h[0] = *(const v8h*)(p);
  f.h[1] = *(const v8h*)(p + 16);
  return f.v;
}

// ------------------------- constants ---------------------------------------
// b=8, SEQ=1280 (padded), DIM=512, HEADS=8, DH=64, text=256, img=1024 (32x32)
#define NB 8
#define TSEQ 1280
#define NDIM 512
#define NHEAD 8
#define DHD 64
#define TEXTL 256
#define IMGW 32
#define BH (NB * NHEAD)

// ------------------- kernel 1: cast + transpose inputs to f16 --------------
// xh     : [NB*TSEQ][512] row-major (padded token 1279 = 0)
// wqkvt  : [1536][512]    = W_qkv^T, q columns pre-scaled by dh^-0.5
// woutt  : [512][512]     = W_out^T
__global__ __launch_bounds__(256) void k_convert(
    const float* __restrict__ x, const float* __restrict__ wqkv,
    const float* __restrict__ wout, _Float16* __restrict__ xh,
    _Float16* __restrict__ wqkvt, _Float16* __restrict__ woutt) {
  const int NX  = NB * TSEQ * NDIM;      // 5242880
  const int NW1 = NDIM * 3 * NDIM;       // 786432
  const int NW2 = NDIM * NDIM;           // 262144
  int idx = blockIdx.x * 256 + threadIdx.x;
  if (idx < NX) {
    int d  = idx & (NDIM - 1);
    int t  = (idx >> 9) % TSEQ;
    int bb = idx / (TSEQ * NDIM);
    float v = (t < TSEQ - 1) ? x[((size_t)bb * (TSEQ - 1) + t) * NDIM + d] : 0.f;
    xh[idx] = (_Float16)v;
  } else if (idx < NX + NW1) {
    int i   = idx - NX;
    int k   = i / (3 * NDIM);
    int col = i - k * (3 * NDIM);
    float v = wqkv[i];
    if (col < NDIM) v *= 0.125f;  // fold q * dh^-0.5 into W_q columns
    wqkvt[(size_t)col * NDIM + k] = (_Float16)v;
  } else if (idx < NX + NW1 + NW2) {
    int i = idx - NX - NW1;
    int k = i >> 9, col = i & (NDIM - 1);
    woutt[(size_t)col * NDIM + k] = (_Float16)wout[i];
  }
}

// ------------------- kernel 2: QKV projection (WMMA) -----------------------
// (NB*TSEQ x 512) @ (512 x 1536). Q,K head-major [hd][t][d];
// V stored TRANSPOSED per head: Vt[hd][d][t] so PV B-fragments are contiguous.
__global__ __launch_bounds__(128) void k_qkv(
    const _Float16* __restrict__ xh, const _Float16* __restrict__ wqkvt,
    _Float16* __restrict__ Qh, _Float16* __restrict__ Kh,
    _Float16* __restrict__ Vt) {
  int lane = threadIdx.x & 31;
  int wave = threadIdx.x >> 5;
  int tile = blockIdx.x * 4 + wave;   // 640 x 96 tiles
  int mt = tile / 96, nt = tile - mt * 96;
  int m0 = mt << 4, n0 = nt << 4;
  v8f acc = {};
  for (int k0 = 0; k0 < NDIM; k0 += 32) {
    v16h a = load_a_rm(xh, NDIM, m0, k0, lane);
    v16h b = load_bt(wqkvt, NDIM, n0, k0, lane);
    acc = wmma32(a, b, acc);
  }
  // tile-uniform decomposition (1280 % 16 == 0: tile never straddles a batch)
  int bb    = mt / 80;                 // batch
  int tbase = (mt - bb * 80) << 4;     // token base
  int Mb = (lane >> 4) << 3;
  int N  = lane & 15;
  int col = n0 + N;
  int s = col >> 9, inner = col & (NDIM - 1);
  int hh = inner >> 6, d = inner & (DHD - 1);
  int hd = bb * NHEAD + hh;
  if (s == 2) {  // V -> transposed layout [hd][d][t]; 8 consecutive tokens
    _Float16* dst = Vt + ((size_t)hd * DHD + d) * TSEQ + tbase + Mb;
#pragma unroll
    for (int v = 0; v < 8; ++v) dst[v] = (_Float16)acc[v];
  } else {
    _Float16* dst = ((s == 0) ? Qh : Kh) + ((size_t)hd * TSEQ) * DHD + d;
#pragma unroll
    for (int v = 0; v < 8; ++v)
      dst[(size_t)(tbase + Mb + v) * DHD] = (_Float16)acc[v];
  }
}

// ------------------- kernel 3: text causal attention -----------------------
// Per wave: one head, 16 query rows; full 16x256 score tile in LDS,
// exact f32 softmax, then P@V with WMMA.
__global__ __launch_bounds__(64) void k_text_attn(
    const _Float16* __restrict__ Qh, const _Float16* __restrict__ Kh,
    const _Float16* __restrict__ Vt, _Float16* __restrict__ AO) {
  __shared__ alignas(16) float    Ssh[2][16][TEXTL];
  __shared__ alignas(16) _Float16 Psh[2][16][TEXTL];
  int lane = threadIdx.x & 31;
  int wave = threadIdx.x >> 5;
  int unit = blockIdx.x * 2 + wave;   // 64 heads * 16 qtiles
  int hd = unit >> 4;
  int qt = unit & 15;
  int m0 = qt << 4;
  const _Float16* Qb  = Qh + (size_t)hd * TSEQ * DHD;
  const _Float16* Kb  = Kh + (size_t)hd * TSEQ * DHD;
  const _Float16* Vtb = Vt + (size_t)hd * DHD * TSEQ;
  int Mb = (lane >> 4) << 3;
  int N  = lane & 15;

  // loop-invariant Q fragments (K-contraction = 64 -> 2 fragments)
  v16h qa0 = load_a_rm(Qb, DHD, m0, 0, lane);
  v16h qa1 = load_a_rm(Qb, DHD, m0, 32, lane);

  // scores S = Q K^T with causal mask
  for (int kt = 0; kt < 16; ++kt) {
    int jmin = kt << 4;
    if (jmin > m0 + 15) {  // fully above diagonal
#pragma unroll
      for (int v = 0; v < 8; ++v) Ssh[wave][Mb + v][jmin + N] = NEGF;
      continue;
    }
    v8f acc = {};
    acc = wmma32(qa0, load_bt(Kb, DHD, jmin, 0, lane), acc);
    acc = wmma32(qa1, load_bt(Kb, DHD, jmin, 32, lane), acc);
#pragma unroll
    for (int v = 0; v < 8; ++v) {
      int i = m0 + Mb + v, j = jmin + N;
      Ssh[wave][Mb + v][j] = (j <= i) ? acc[v] : NEGF;
    }
  }
  __syncthreads();
  if (lane < 16) {  // one row per lane: exact softmax in f32
    int r = lane;
    float m = NEGF;
    for (int j = 0; j < TEXTL; ++j) m = fmaxf(m, Ssh[wave][r][j]);
    float s = 0.f;
    for (int j = 0; j < TEXTL; ++j) s += __expf(Ssh[wave][r][j] - m);
    float inv = 1.f / s;
    for (int j = 0; j < TEXTL; ++j)
      Psh[wave][r][j] = (_Float16)(__expf(Ssh[wave][r][j] - m) * inv);
  }
  __syncthreads();
  // O = P V   (skip key blocks that are fully masked)
  v8f o0 = {}, o1 = {}, o2 = {}, o3 = {};
  int kbmax = (m0 + 15) >> 5;
  for (int kb = 0; kb <= kbmax; ++kb) {
    v16h a = load_a_rm(&Psh[wave][0][0], TEXTL, 0, kb << 5, lane);
    o0 = wmma32(a, load_bt(Vtb, TSEQ, 0, kb << 5, lane), o0);
    o1 = wmma32(a, load_bt(Vtb, TSEQ, 16, kb << 5, lane), o1);
    o2 = wmma32(a, load_bt(Vtb, TSEQ, 32, kb << 5, lane), o2);
    o3 = wmma32(a, load_bt(Vtb, TSEQ, 48, kb << 5, lane), o3);
  }
  int bb = hd >> 3, h = hd & 7;
  v8f oo[4] = {o0, o1, o2, o3};
#pragma unroll
  for (int nt = 0; nt < 4; ++nt)
#pragma unroll
    for (int v = 0; v < 8; ++v) {
      int t = m0 + Mb + v;
      AO[((size_t)(bb * TSEQ + t)) * NDIM + h * DHD + (nt << 4) + N] =
          (_Float16)oo[nt][v];
    }
}

// ------------------- kernel 4: axial image attention -----------------------
// Per wave: one head, one image row x, 16 of its 32 queries.
// Keys: 256 text tokens (masked) + 32 causal in-row image tokens.
__global__ __launch_bounds__(64) void k_img_attn(
    const _Float16* __restrict__ Qh, const _Float16* __restrict__ Kh,
    const _Float16* __restrict__ Vt, const unsigned char* __restrict__ mask,
    _Float16* __restrict__ AO) {
  __shared__ alignas(16) float    Ssh[2][16][TEXTL + IMGW];
  __shared__ alignas(16) _Float16 Psh[2][16][TEXTL + IMGW];
  const int PITCH = TEXTL + IMGW;  // 288
  int lane = threadIdx.x & 31;
  int wave = threadIdx.x >> 5;
  int unit = blockIdx.x * 2 + wave;  // 64 heads * 32 rows * 2 halves
  int hd = unit >> 6;
  int rem = unit & 63;
  int x = rem >> 1;
  int half = rem & 1;
  int qtok = TEXTL + x * IMGW + (half << 4);  // first query token
  const _Float16* Qb  = Qh + (size_t)hd * TSEQ * DHD;
  const _Float16* Kb  = Kh + (size_t)hd * TSEQ * DHD;
  const _Float16* Vtb = Vt + (size_t)hd * DHD * TSEQ;
  int bb = hd >> 3, h = hd & 7;
  int Mb = (lane >> 4) << 3;
  int N  = lane & 15;

  v16h qa0 = load_a_rm(Qb, DHD, qtok, 0, lane);
  v16h qa1 = load_a_rm(Qb, DHD, qtok, 32, lane);

  // text-key tiles (mask applies; all-ones in this workload)
  for (int kt = 0; kt < 16; ++kt) {
    int jmin = kt << 4;
    v8f acc = {};
    acc = wmma32(qa0, load_bt(Kb, DHD, jmin, 0, lane), acc);
    acc = wmma32(qa1, load_bt(Kb, DHD, jmin, 32, lane), acc);
    int j = jmin + N;
    bool ok = mask[bb * TEXTL + j] != 0;
#pragma unroll
    for (int v = 0; v < 8; ++v)
      Ssh[wave][Mb + v][j] = ok ? acc[v] : NEGF;
  }
  // in-row image-key tiles (causal within the row)
  for (int kt = 0; kt < 2; ++kt) {
    int jc0 = kt << 4;
    if (jc0 > (half << 4) + 15) {  // fully above diagonal
#pragma unroll
      for (int v = 0; v < 8; ++v) Ssh[wave][Mb + v][TEXTL + jc0 + N] = NEGF;
      continue;
    }
    v8f acc = {};
    acc = wmma32(qa0, load_bt(Kb, DHD, TEXTL + x * IMGW + jc0, 0, lane), acc);
    acc = wmma32(qa1, load_bt(Kb, DHD, TEXTL + x * IMGW + jc0, 32, lane), acc);
#pragma unroll
    for (int v = 0; v < 8; ++v) {
      int c = (half << 4) + Mb + v;  // query column within row
      int jc = jc0 + N;
      Ssh[wave][Mb + v][TEXTL + jc] = (jc <= c) ? acc[v] : NEGF;
    }
  }
  __syncthreads();
  if (lane < 16) {
    int r = lane;
    float m = NEGF;
    for (int j = 0; j < PITCH; ++j) m = fmaxf(m, Ssh[wave][r][j]);
    float s = 0.f;
    for (int j = 0; j < PITCH; ++j) s += __expf(Ssh[wave][r][j] - m);
    float inv = 1.f / s;
    for (int j = 0; j < PITCH; ++j)
      Psh[wave][r][j] = (_Float16)(__expf(Ssh[wave][r][j] - m) * inv);
  }
  __syncthreads();
  v8f o0 = {}, o1 = {}, o2 = {}, o3 = {};
  for (int kb = 0; kb < 8; ++kb) {  // text value blocks
    v16h a = load_a_rm(&Psh[wave][0][0], PITCH, 0, kb << 5, lane);
    o0 = wmma32(a, load_bt(Vtb, TSEQ, 0, kb << 5, lane), o0);
    o1 = wmma32(a, load_bt(Vtb, TSEQ, 16, kb << 5, lane), o1);
    o2 = wmma32(a, load_bt(Vtb, TSEQ, 32, kb << 5, lane), o2);
    o3 = wmma32(a, load_bt(Vtb, TSEQ, 48, kb << 5, lane), o3);
  }
  {  // in-row image value block (P cols 256..287)
    int vtok = TEXTL + x * IMGW;
    v16h a = load_a_rm(&Psh[wave][0][0], PITCH, 0, TEXTL, lane);
    o0 = wmma32(a, load_bt(Vtb, TSEQ, 0, vtok, lane), o0);
    o1 = wmma32(a, load_bt(Vtb, TSEQ, 16, vtok, lane), o1);
    o2 = wmma32(a, load_bt(Vtb, TSEQ, 32, vtok, lane), o2);
    o3 = wmma32(a, load_bt(Vtb, TSEQ, 48, vtok, lane), o3);
  }
  v8f oo[4] = {o0, o1, o2, o3};
#pragma unroll
  for (int nt = 0; nt < 4; ++nt)
#pragma unroll
    for (int v = 0; v < 8; ++v) {
      int t = qtok + Mb + v;
      AO[((size_t)(bb * TSEQ + t)) * NDIM + h * DHD + (nt << 4) + N] =
          (_Float16)oo[nt][v];
    }
}

// ------------------- kernel 5: output projection ---------------------------
__global__ __launch_bounds__(128) void k_outproj(
    const _Float16* __restrict__ AO, const _Float16* __restrict__ woutt,
    const float* __restrict__ b_out, float* __restrict__ out) {
  int lane = threadIdx.x & 31;
  int wave = threadIdx.x >> 5;
  int tile = blockIdx.x * 4 + wave;  // 640 x 32 tiles
  int mt = tile >> 5, nt = tile & 31;
  int m0 = mt << 4, n0 = nt << 4;
  v8f acc = {};
  for (int k0 = 0; k0 < NDIM; k0 += 32) {
    v16h a = load_a_rm(AO, NDIM, m0, k0, lane);
    v16h b = load_bt(woutt, NDIM, n0, k0, lane);
    acc = wmma32(a, b, acc);
  }
  int bb    = mt / 80;
  int tbase = (mt - bb * 80) << 4;
  int Mb = (lane >> 4) << 3;
  int N  = lane & 15;
  int col = n0 + N;
  float bias = b_out[col];
#pragma unroll
  for (int v = 0; v < 8; ++v) {
    int t = tbase + Mb + v;
    if (t < TSEQ - 1)
      out[((size_t)bb * (TSEQ - 1) + t) * NDIM + col] = acc[v] + bias;
  }
}

// ---------------------------------------------------------------------------
extern "C" void kernel_launch(void* const* d_in, const int* in_sizes, int n_in,
                              void* d_out, int out_size, void* d_ws,
                              size_t ws_size, hipStream_t stream) {
  (void)in_sizes; (void)n_in; (void)out_size; (void)ws_size;
  const float* x            = (const float*)d_in[0];
  const unsigned char* mask = (const unsigned char*)d_in[1];
  const float* wqkv         = (const float*)d_in[2];
  const float* wout         = (const float*)d_in[3];
  const float* bout         = (const float*)d_in[4];
  float* out                = (float*)d_out;
  char* ws                  = (char*)d_ws;

  const size_t SZ_XH   = (size_t)NB * TSEQ * NDIM * 2;      // 10,485,760
  const size_t SZ_WQKV = (size_t)NDIM * 3 * NDIM * 2;       //  1,572,864
  const size_t SZ_WOUT = (size_t)NDIM * NDIM * 2;           //    524,288
  const size_t SZ_HEAD = (size_t)BH * TSEQ * DHD * 2;       // 10,485,760

  _Float16* xh    = (_Float16*)(ws);
  _Float16* wqkvt = (_Float16*)(ws + SZ_XH);
  _Float16* woutt = (_Float16*)(ws + SZ_XH + SZ_WQKV);
  _Float16* Qh    = (_Float16*)(ws + SZ_XH + SZ_WQKV + SZ_WOUT);
  _Float16* Kh    = (_Float16*)(ws + SZ_XH + SZ_WQKV + SZ_WOUT + SZ_HEAD);
  _Float16* Vt    = (_Float16*)(ws + SZ_XH + SZ_WQKV + SZ_WOUT + 2 * SZ_HEAD);
  _Float16* AO    = (_Float16*)(ws + SZ_XH + SZ_WQKV + SZ_WOUT + 3 * SZ_HEAD);

  // 1) cast + pad x, transpose+cast weights (q-scale folded into W_qkv)
  k_convert<<<(NB * TSEQ * NDIM + NDIM * 3 * NDIM + NDIM * NDIM + 255) / 256,
              256, 0, stream>>>(x, wqkv, wout, xh, wqkvt, woutt);
  // 2) QKV projection: 640x96 16x16 tiles, 4 waves/block
  k_qkv<<<(640 * 96) / 4, 128, 0, stream>>>(xh, wqkvt, Qh, Kh, Vt);
  // 3) text attention: 64 heads * 16 query tiles, 2 waves/block
  k_text_attn<<<(BH * 16) / 2, 64, 0, stream>>>(Qh, Kh, Vt, AO);
  // 4) image axial attention: 64 heads * 32 rows * 2 halves, 2 waves/block
  k_img_attn<<<(BH * IMGW * 2) / 2, 64, 0, stream>>>(Qh, Kh, Vt, mask, AO);
  // 5) output projection: 640x32 tiles, 4 waves/block
  k_outproj<<<(640 * 32) / 4, 128, 0, stream>>>(AO, woutt, bout, out);
}
